// SDPA_19610820674059
// MI455X (gfx1250) — matrix operand order repaired and checked by reading, not compiled, not run on hardware
//
#include <hip/hip_runtime.h>
#include <hip/hip_bf16.h>

typedef float    v8f  __attribute__((ext_vector_type(8)));
typedef float    v4f  __attribute__((ext_vector_type(4)));
typedef _Float16 v16h __attribute__((ext_vector_type(16)));
typedef _Float16 v8h  __attribute__((ext_vector_type(8)));
typedef _Float16 v2h  __attribute__((ext_vector_type(2)));
typedef unsigned v8u  __attribute__((ext_vector_type(8)));

#define LQ    2048      // L == S
#define DQ    64
#define BH    16        // B*H
#define EQ    131072    // 2048*64 elements per (b,h)
#define NOUT1 2097152   // 2*8*2048*64
// alpha = sqrt(0.125 * log2(e)); scores from (alpha*Q)(alpha*K)^T feed v_exp_f32
// (= exp2) directly: exp2(alpha^2 * qk) = exp(qk/8). No VALU scaling needed.
#define ALPHA 0.42466089f

static __device__ inline v8h ld8(const _Float16* __restrict__ p) {
    return *(const v8h* __restrict__)p;
}
static __device__ inline v16h hcat(v8h lo, v8h hi) {
    return __builtin_shufflevector(lo, hi, 0,1,2,3,4,5,6,7,8,9,10,11,12,13,14,15);
}
static __device__ inline v8f wmma16(v16h a, v16h b, v8f c) {
    return __builtin_amdgcn_wmma_f32_16x16x32_f16(false, a, false, b, (short)0, c, false, false);
}
static __device__ inline v8f zero8() {
    v8f z = {0.f,0.f,0.f,0.f,0.f,0.f,0.f,0.f}; return z;
}
static __device__ inline float e2(float x) {          // v_exp_f32 (base-2)
    return __builtin_amdgcn_exp2f(x);
}
// swap lanes l <-> l^16 (group-of-32 swizzle, SWAPX16: xor=0x10,and=0x1f)
static __device__ inline unsigned swz16u(unsigned x) {
    return (unsigned)__builtin_amdgcn_ds_swizzle((int)x, 0x401F);
}
static __device__ inline float swz16f(float x) {
    return __int_as_float(__builtin_amdgcn_ds_swizzle(__float_as_int(x), 0x401F));
}
static __device__ inline unsigned pkh2(float a, float b) {   // v_cvt_pkrtz_f16_f32
    return __builtin_bit_cast(unsigned, __builtin_amdgcn_cvt_pkrtz(a, b));
}
static __device__ inline float sumexp8(v8f c) {
    float s = 0.f;
#pragma unroll
    for (int j = 0; j < 8; ++j) s += e2(c[j]);
    return s;
}

// ---- kernel 1: f32 -> f16 conversion (Q,K pre-scaled by ALPHA), V transpose ----
__global__ __launch_bounds__(256) void SDPA_cvt(
    const float* __restrict__ q, const float* __restrict__ k, const float* __restrict__ v,
    _Float16* __restrict__ qh, _Float16* __restrict__ kh, _Float16* __restrict__ vth)
{
    int i = blockIdx.x * 256 + threadIdx.x;          // < 2,097,152
    qh[i] = (_Float16)(q[i] * ALPHA);
    kh[i] = (_Float16)(k[i] * ALPHA);
    int bh = i >> 17, rem = i & (EQ - 1);
    int s = rem >> 6, d = rem & 63;
    vth[(size_t)bh * EQ + (size_t)d * LQ + s] = (_Float16)v[i];
}

// ---- kernel 2: fused SDPA + self-similarity softmaxes ----
// grid: 16 bh * 128 query-blocks; 8 waves, each owns an s-range of 256.
__global__ __launch_bounds__(256) void SDPA_main(
    const _Float16* __restrict__ qh, const _Float16* __restrict__ kh,
    const _Float16* __restrict__ vth, float* __restrict__ out1, float* __restrict__ out2)
{
    const int tid  = threadIdx.x;
    const int wave = tid >> 5, lane = tid & 31;
    const int lh = lane & 15, hs = lane >> 4;
    const int bh = blockIdx.x >> 7, lblk = blockIdx.x & 127, l0 = lblk << 4;

    const _Float16* __restrict__ Q  = qh  + (size_t)bh * EQ;
    const _Float16* __restrict__ K  = kh  + (size_t)bh * EQ;
    const _Float16* __restrict__ VT = vth + (size_t)bh * EQ;

    // B-layout (32x16) operands for the 16 queries/keys of this block:
    // lane: n = l0+lh, halves = 16 contiguous d at 32*c + 16*hs
    const _Float16* qrow = Q + (size_t)(l0 + lh) * DQ;
    const _Float16* krow = K + (size_t)(l0 + lh) * DQ;
    v16h Bq0 = hcat(ld8(qrow + 16*hs),      ld8(qrow + 16*hs + 8));
    v16h Bq1 = hcat(ld8(qrow + 32 + 16*hs), ld8(qrow + 32 + 16*hs + 8));
    v16h Bk0 = hcat(ld8(krow + 16*hs),      ld8(krow + 16*hs + 8));
    v16h Bk1 = hcat(ld8(krow + 32 + 16*hs), ld8(krow + 32 + 16*hs + 8));

    // A-layout (16x32) loader: lane row = r0+lh, halves = 16 contiguous at 16*hs per 32-chunk
    auto loadA = [&](const _Float16* __restrict__ M, int r0, int c0) -> v16h {
        const _Float16* row = M + (size_t)(r0 + lh) * DQ + c0;
        return hcat(ld8(row + 16*hs), ld8(row + 16*hs + 8));
    };

    const int sbase = wave << 8;   // 256 columns per wave

    // ------- pass 1: exp row-sums (no max subtraction: |score| bounded) -----
    float sqk = 0.f, sqq = 0.f, skk = 0.f;
    for (int t = 0; t < 16; ++t) {
        int s0 = sbase + (t << 4);
        v16h Ak0 = loadA(K, s0, 0), Ak1 = loadA(K, s0, 32);
        v16h Aq0 = loadA(Q, s0, 0), Aq1 = loadA(Q, s0, 32);
        v8f c = zero8(); c = wmma16(Ak0, Bq0, c); c = wmma16(Ak1, Bq1, c);  // (K Qt)[s][l]
        sqk += sumexp8(c);
        v8f c2 = zero8(); c2 = wmma16(Aq0, Bq0, c2); c2 = wmma16(Aq1, Bq1, c2); // QQt
        sqq += sumexp8(c2);
        v8f c3 = zero8(); c3 = wmma16(Ak0, Bk0, c3); c3 = wmma16(Ak1, Bk1, c3); // KKt
        skk += sumexp8(c3);
    }
    sqk += swz16f(sqk); sqq += swz16f(sqq); skk += swz16f(skk);

    __shared__ float red[3][8][16];
    if (hs == 0) { red[0][wave][lh] = sqk; red[1][wave][lh] = sqq; red[2][wave][lh] = skk; }
    __syncthreads();
    float tq = 0.f, tg = 0.f, tk = 0.f;
#pragma unroll
    for (int w = 0; w < 8; ++w) { tq += red[0][w][lh]; tg += red[1][w][lh]; tk += red[2][w][lh]; }
    const float rq = 1.f / tq, rqq = 1.f / tg, rkk = 1.f / tk;

    // ------- pass 2: recompute, write out2, accumulate (unnormalized) P.V ---
    v8f o[4] = { zero8(), zero8(), zero8(), zero8() };   // outT accum, d-blocks
    float* o2row = out2 + ((size_t)(bh * LQ + l0 + lh)) * LQ;

    for (int tp = 0; tp < 8; ++tp) {
        int s0 = sbase + (tp << 5);
        unsigned pe[4], pex[4], po[4], pox[4];
#pragma unroll
        for (int half = 0; half < 2; ++half) {
            int st = s0 + 16 * half;
            v16h Ak0 = loadA(K, st, 0), Ak1 = loadA(K, st, 32);
            v16h Aq0 = loadA(Q, st, 0), Aq1 = loadA(Q, st, 32);
            v8f c  = zero8(); c  = wmma16(Ak0, Bq0, c);  c  = wmma16(Ak1, Bq1, c);
            v8f cq = zero8(); cq = wmma16(Aq0, Bq0, cq); cq = wmma16(Aq1, Bq1, cq);
            v8f ck = zero8(); ck = wmma16(Ak0, Bk0, ck); ck = wmma16(Ak1, Bk1, ck);
            float p[8]; v4f wlo, whi;
#pragma unroll
            for (int j = 0; j < 8; ++j) {
                p[j] = e2(c[j]);                                  // unnormalized
                float w2 = fmaf(e2(cq[j]), rqq, e2(ck[j]) * rkk); // out2 element
                if (j < 4) wlo[j] = w2; else whi[j - 4] = w2;
            }
            // out2[l][m], 8 consecutive m per lane (streaming: NT stores)
            __builtin_nontemporal_store(wlo, (v4f*)(o2row + st + 8 * hs));
            __builtin_nontemporal_store(whi, (v4f*)(o2row + st + 8 * hs + 4));
            // pack Pt (v_cvt_pkrtz) and pre-swizzle for B-layout assembly
#pragma unroll
            for (int j = 0; j < 4; ++j) {
                unsigned u = pkh2(p[2 * j], p[2 * j + 1]);
                unsigned x = swz16u(u);
                if (half == 0) { pe[j] = u; pex[j] = x; }
                else           { po[j] = u; pox[j] = x; }
            }
        }
        // Assemble Pt (32x16) in B layout: lanes 0-15 <- even tile, 16-31 <- odd
        v8u bu;
#pragma unroll
        for (int j = 0; j < 4; ++j) {
            bu[j]     = hs ? pox[j] : pe[j];
            bu[j + 4] = hs ? po[j]  : pex[j];
        }
        v16h Bp = __builtin_bit_cast(v16h, bu);
        // outT[d][l] += VT[d][s-chunk] x Pt[s][l]
#pragma unroll
        for (int blk = 0; blk < 4; ++blk) {
            const _Float16* vrow = VT + (size_t)(16 * blk + lh) * LQ + s0;
            v16h Av = hcat(ld8(vrow + 16 * hs), ld8(vrow + 16 * hs + 8));
            o[blk] = wmma16(Av, Bp, o[blk]);
        }
    }

    // ------- cross-wave P.V reduction, deferred 1/sum scale, out1 write -----
    __shared__ float pvred[8][4][8][32];   // 32 KB
#pragma unroll
    for (int blk = 0; blk < 4; ++blk)
#pragma unroll
        for (int vg = 0; vg < 8; ++vg)
            pvred[wave][blk][vg][lane] = o[blk][vg];
    __syncthreads();
    if (wave < 4) {
        const int blk = wave;
        float acc[8];
#pragma unroll
        for (int vg = 0; vg < 8; ++vg) {
            float a = 0.f;
#pragma unroll
            for (int w = 0; w < 8; ++w) a += pvred[w][blk][vg][lane];
            acc[vg] = a * rq;                       // softmax normalization here
        }
        float* orow = out1 + ((size_t)(bh * LQ + l0 + lh)) * DQ + 16 * blk + 8 * hs;
        v4f lo = { acc[0], acc[1], acc[2], acc[3] };
        v4f hi = { acc[4], acc[5], acc[6], acc[7] };
        *(v4f*)orow       = lo;
        *(v4f*)(orow + 4) = hi;
    }
}

extern "C" void kernel_launch(void* const* d_in, const int* in_sizes, int n_in,
                              void* d_out, int out_size, void* d_ws, size_t ws_size,
                              hipStream_t stream) {
    const float* q = (const float*)d_in[0];
    const float* k = (const float*)d_in[1];
    const float* v = (const float*)d_in[2];

    _Float16* wsh = (_Float16*)d_ws;       // needs 3 * 2,097,152 * 2B = 12 MB
    _Float16* qh  = wsh;
    _Float16* kh  = wsh + (size_t)BH * EQ;
    _Float16* vth = wsh + 2 * (size_t)BH * EQ;

    float* out1 = (float*)d_out;
    float* out2 = out1 + NOUT1;

    SDPA_cvt<<<(BH * EQ) / 256, 256, 0, stream>>>(q, k, v, qh, kh, vth);
    SDPA_main<<<BH * 128, 256, 0, stream>>>(qh, kh, vth, out1, out2);
}
